// SCLDecoder_22686017257977
// MI455X (gfx1250) — compile-verified, matrix operand order checked by reading, not run on hardware
//
#include <hip/hip_runtime.h>
#include <cstddef>

// ---------------------------------------------------------------------------
// CDNA5 (gfx1250) neural SCL polar decoder.
//   - All GEMMs use v_wmma_f32_16x16x32_f16 (wave32, 16x16 tiles, f32 accum).
//   - Weights pre-packed into WMMA B-fragment layout once per call.
//   - DFS over the code tree is enqueued host-side as a fixed kernel chain.
// ---------------------------------------------------------------------------

typedef __attribute__((ext_vector_type(16))) _Float16 v16h;
typedef __attribute__((ext_vector_type(8)))  float    v8f;
typedef __attribute__((ext_vector_type(4)))  unsigned int u32x4;

union AFrag { v16h v; u32x4 q[2]; };

#define BATCH 128
#define NLIST 8
#define NBITS 64
#define DEMB  64
#define HDIM  256
#define BL    (BATCH * NLIST)   // 1024

// ---------------------------------------------------------------------------
// Generic WMMA GEMM: Out(M x Ncols) = X(M x Kpad,f16) * Bp(packed f16) + bias
// One wave per 16-row stripe. A fragments cached in VGPRs across the n loop.
// A layout (16x32 f16, M x K): lane m=L%16, hi=L/16; halves j<8 -> k=kt*32+hi*8+j
//                              halves j>=8 -> k=kt*32+16+hi*8+(j-8)   (contig 16B)
// B packed so each lane's 16 halves are contiguous (32B) in memory.
// D layout: vgpr r -> row m0 + hi*8 + r, col nt*16 + (L&15).
// ---------------------------------------------------------------------------
template <int KT, bool RELU_F16OUT>
__global__ __launch_bounds__(32)
void gemm_wmma(const _Float16* __restrict__ Xm, const _Float16* __restrict__ Bp,
               const float* __restrict__ bias, void* __restrict__ Out,
               int Kpad, int Ncols)
{
    const int lane = threadIdx.x;
    const int hi   = lane >> 4;
    const int m0   = blockIdx.x * 16;
    const int mrow = m0 + (lane & 15);

    AFrag a[KT];
    const _Float16* xr = Xm + (size_t)mrow * Kpad;
#pragma unroll
    for (int kt = 0; kt < KT; ++kt) {
        a[kt].q[0] = *reinterpret_cast<const u32x4*>(xr + kt * 32 + hi * 8);
        a[kt].q[1] = *reinterpret_cast<const u32x4*>(xr + kt * 32 + 16 + hi * 8);
    }

    const int NT = Ncols >> 4;
    for (int nt = 0; nt < NT; ++nt) {
        v8f acc = {0.f, 0.f, 0.f, 0.f, 0.f, 0.f, 0.f, 0.f};
#pragma unroll
        for (int kt = 0; kt < KT; ++kt) {
            AFrag b;
            const _Float16* bp = Bp + (((size_t)(kt * NT + nt) * 32) + lane) * 16;
            b.q[0] = *reinterpret_cast<const u32x4*>(bp);
            b.q[1] = *reinterpret_cast<const u32x4*>(bp + 8);
            acc = __builtin_amdgcn_wmma_f32_16x16x32_f16(
                false, a[kt].v, false, b.v, (short)0, acc, false, false);
        }
        const int n  = nt * 16 + (lane & 15);
        const float bs = bias ? bias[n] : 0.f;
        if (RELU_F16OUT) {
            _Float16* o = (_Float16*)Out;
#pragma unroll
            for (int r = 0; r < 8; ++r) {
                float v = acc[r] + bs;
                v = v > 0.f ? v : 0.f;
                o[(size_t)(m0 + hi * 8 + r) * Ncols + n] = (_Float16)v;
            }
        } else {
            float* o = (float*)Out;
#pragma unroll
            for (int r = 0; r < 8; ++r)
                o[(size_t)(m0 + hi * 8 + r) * Ncols + n] = acc[r] + bs;
        }
    }
}

// Pack W (Krows x Ncols row-major f32) into WMMA B-fragment order (f16).
__global__ void pack_b(const float* __restrict__ W, _Float16* __restrict__ P,
                       int Kreal, int KT, int NT, int Ncols)
{
    int idx = blockIdx.x * blockDim.x + threadIdx.x;
    int total = KT * NT * 32 * 16;
    if (idx >= total) return;
    int j    = idx & 15;
    int lane = (idx >> 4) & 31;
    int t    = idx >> 9;
    int nt   = t % NT;
    int kt   = t / NT;
    int n    = nt * 16 + (lane & 15);
    int hi   = lane >> 4;
    int k    = kt * 32 + ((j < 8) ? (hi * 8 + j) : (16 + hi * 8 + (j - 8)));
    float v  = (k < Kreal) ? W[(size_t)k * Ncols + n] : 0.f;
    P[idx]   = (_Float16)v;
}

// Obs MLP layer-1 (K=4 is too small for WMMA): plain FMA, relu, f16 out.
__global__ void obs_mlp1(const float* __restrict__ y, const float* __restrict__ W1,
                         const float* __restrict__ b1, _Float16* __restrict__ H, int M)
{
    int idx = blockIdx.x * blockDim.x + threadIdx.x;
    if (idx >= M * HDIM) return;
    int m = idx >> 8, c = idx & 255;
    const float* yr = y + (size_t)m * 4;
    float v = b1[c];
#pragma unroll
    for (int k = 0; k < 4; ++k) v += yr[k] * W1[k * HDIM + c];
    H[idx] = (_Float16)(v > 0.f ? v : 0.f);
}

// Broadcast e0 (B,N,DE) across the list dimension -> e[0] (B,nL,N,DE)
__global__ void bcast_e0(const float* __restrict__ e0, float* __restrict__ e, int total)
{
    int idx = blockIdx.x * blockDim.x + threadIdx.x;
    if (idx >= total) return;
    int d = idx & 63;
    int n = (idx >> 6) & 63;
    int b = idx >> 15;           // skip l = (idx>>12)&7
    e[idx] = e0[(((size_t)b * 64 + n) << 6) + d];
}

__global__ void init_pm(float* pm)
{
    int idx = blockIdx.x * blockDim.x + threadIdx.x;
    if (idx >= BL) return;
    pm[idx] = ((idx & 7) == 0) ? 0.f : 1e8f;
}

// chk concat: X[r][0:64]=e[..,2k,:], X[r][64:128]=e[..,2k+1,:]  (f32 -> f16)
__global__ void concat_chk(const float* __restrict__ e, _Float16* __restrict__ X,
                           int half, int total)
{
    int idx = blockIdx.x * blockDim.x + threadIdx.x;
    if (idx >= total) return;
    int c = idx & 127;
    int r = idx >> 7;
    int k  = r % half;
    int bl = r / half;
    int sn = 2 * k + (c >= 64 ? 1 : 0);
    X[(size_t)r * 128 + c] =
        (_Float16)e[(((size_t)bl * (2 * half) + sn) << 6) + (c & 63)];
}

// bit concat (K padded 144->160): gathered e_odd|e_even (by ord), u-embedding, zero pad.
__global__ void concat_bit(const float* __restrict__ e, const int* __restrict__ ord,
                           const float* __restrict__ u1hp, const float* __restrict__ emb,
                           _Float16* __restrict__ X, int half, int total)
{
    int idx = blockIdx.x * blockDim.x + threadIdx.x;
    if (idx >= total) return;
    int c = idx % 160;
    int r = idx / 160;
    int k  = r % half;
    int bl = r / half;
    int b  = bl >> 3;
    int l  = bl & 7;
    float v;
    if (c < 128) {
        int lg = ord[b * NLIST + l];
        int sn = 2 * k + (c >= 64 ? 1 : 0);
        v = e[(((size_t)(b * NLIST + lg) * (2 * half) + sn) << 6) + (c & 63)];
    } else if (c < 144) {
        int u1 = (int)u1hp[(size_t)bl * half + k];
        v = emb[u1 * 16 + (c - 128)];
    } else {
        v = 0.f;
    }
    X[(size_t)r * 160 + c] = (_Float16)v;
}

// Leaf (N==1): LLR head, path-metric update, list prune/reorder. One wave per batch.
__global__ __launch_bounds__(32)
void leaf_kernel(const float* __restrict__ e6, const int* __restrict__ f, int bit,
                 float* __restrict__ pm, float* __restrict__ uhat6, float* __restrict__ x6,
                 float* __restrict__ llr6, int* __restrict__ ord6,
                 const float* __restrict__ Wllr, const float* __restrict__ bllr)
{
    const int b = blockIdx.x;
    const int lane = threadIdx.x;
    __shared__ float llr_s[NLIST];
    __shared__ int   hd_s[NLIST];
    __shared__ float pmn[NLIST], xs[NLIST];
    __shared__ int   os[NLIST];

    if (lane < NLIST) {
        const float* er = e6 + (((size_t)b * NLIST + lane) << 6);
        float t = bllr[1] - bllr[0];
        for (int d = 0; d < DEMB; ++d) t += er[d] * (Wllr[d * 2 + 1] - Wllr[d * 2]);
        // clip(sigmoid(t), EPS, 1-EPS) -> logit == clamp(t, +-log((1-EPS)/EPS))
        const float L = 13.815509557963774f;
        float llr = t;
        if (llr > L) llr = L;
        if (llr < -L) llr = -L;
        llr_s[lane] = llr;
        hd_s[lane]  = (t > 0.f) ? 1 : 0;
    }
    __syncthreads();
    if (lane == 0) {
        float pmd[16];
        int   hd16[16];
        for (int i = 0; i < 8; ++i) { pmd[i] = pm[b * NLIST + i]; hd16[i] = hd_s[i]; }
        for (int i = 0; i < 8; ++i) {
            pmd[8 + i]  = pm[b * NLIST + i] + fabsf(llr_s[i]);
            hd16[8 + i] = 1 - hd_s[i];
        }
        int fb = f[(size_t)b * NBITS + bit];
        if (fb != 2) {                       // frozen bit
            for (int l = 0; l < 8; ++l) {
                xs[l]  = (float)fb;
                float mism = (hd_s[l] != fb) ? 1.f : 0.f;
                pmn[l] = pm[b * NLIST + l] + fabsf(llr_s[l]) * mism;
                os[l]  = l;
            }
        } else {                             // info bit: keep 8 smallest of 16 (stable)
            int cnt = 0;
            for (int i = 0; i < 16 && cnt < 8; ++i) {
                int c = 0;
                for (int j = 0; j < 16; ++j)
                    if (pmd[j] < pmd[i] || (pmd[j] == pmd[i] && j < i)) c++;
                if (c < 8) { xs[cnt] = (float)hd16[i]; pmn[cnt] = pmd[i]; os[cnt] = i & 7; cnt++; }
            }
        }
    }
    __syncthreads();
    if (lane < NLIST) {
        int o = b * NLIST + lane;
        uhat6[o] = xs[lane];
        x6[o]    = xs[lane];
        llr6[o]  = llr_s[lane];
        pm[o]    = pmn[lane];
        ord6[o]  = os[lane];
    }
}

// Combine after the right child: gather-left-by-ord2, concat u/llr, xor partial sums.
__global__ void combine_kernel(int half, const int* __restrict__ ord2,
                               const float* __restrict__ uhat_sv, const float* __restrict__ llr_sv,
                               const float* __restrict__ x_sv,   const int* __restrict__ ordL_sv,
                               const float* __restrict__ uhat_c, const float* __restrict__ llr_c,
                               const float* __restrict__ x_c,
                               float* __restrict__ uhat_p, float* __restrict__ llr_p,
                               float* __restrict__ x_p, int* __restrict__ ord_out, int total)
{
    int idx = blockIdx.x * blockDim.x + threadIdx.x;
    if (idx >= total) return;
    int k  = idx % half;
    int bl = idx / half;
    int b  = bl >> 3;
    int l  = bl & 7;
    int o2 = ord2[b * NLIST + l];
    int sbl = b * NLIST + o2;
    int Nn  = 2 * half;
    uhat_p[(size_t)bl * Nn + k]        = uhat_sv[(size_t)sbl * half + k];
    uhat_p[(size_t)bl * Nn + half + k] = uhat_c[(size_t)bl * half + k];
    llr_p[(size_t)bl * Nn + k]         = llr_sv[(size_t)sbl * half + k];
    llr_p[(size_t)bl * Nn + half + k]  = llr_c[(size_t)bl * half + k];
    float u1 = x_sv[(size_t)sbl * half + k];
    float u2 = x_c[(size_t)bl * half + k];
    x_p[(size_t)bl * Nn + 2 * k]     = (float)(((int)u1 + (int)u2) & 1);
    x_p[(size_t)bl * Nn + 2 * k + 1] = u2;
    if (k == 0) ord_out[b * NLIST + l] = ordL_sv[b * NLIST + o2];
}

// Output: best-list uhat (B*N floats), then full-list llr (B*nL*N floats).
__global__ void final_kernel(const float* __restrict__ pm, const float* __restrict__ uhat0,
                             const float* __restrict__ llr0, float* __restrict__ out,
                             int out_size)
{
    int idx = blockIdx.x * blockDim.x + threadIdx.x;
    const int total = BATCH * NBITS + BL * NBITS;
    if (idx >= total || idx >= out_size) return;
    if (idx < BATCH * NBITS) {
        int b = idx / NBITS, i = idx % NBITS;
        int best = 0;
        float mv = pm[b * NLIST];
        for (int l = 1; l < NLIST; ++l) {
            float v = pm[b * NLIST + l];
            if (v < mv) { mv = v; best = l; }
        }
        out[idx] = uhat0[((size_t)b * NLIST + best) * NBITS + i];
    } else {
        out[idx] = llr0[idx - BATCH * NBITS];
    }
}

// ---------------------------------------------------------------------------
// Host-side driver: fixed DFS enqueue (deterministic, graph-capture safe).
// ---------------------------------------------------------------------------
namespace {

struct Ctx {
    hipStream_t s;
    float *e[7], *uhat[7], *xb[7], *llr[7];
    float *uhat_sv[6], *llr_sv[6], *x_sv[6];
    int   *ordL_sv[6], *ord_ret[7];
    float *pm;
    _Float16 *X, *H;
    _Float16 *Pchk1, *Pchk2, *Pbit1, *Pbit2, *Pobs2;
    const float *b_chk1, *b_chk2, *b_bit1, *b_bit2;
    const float *emb, *Wllr, *bllr;
    const int *f;
};

static void decode_rec(Ctx& c, int lvl, int bit, int Nn)
{
    if (Nn == 1) {
        leaf_kernel<<<BATCH, 32, 0, c.s>>>(c.e[lvl], c.f, bit, c.pm,
                                           c.uhat[lvl], c.xb[lvl], c.llr[lvl],
                                           c.ord_ret[lvl], c.Wllr, c.bllr);
        return;
    }
    const int half = Nn / 2;
    const int M    = BL * half;

    // ---- check-node MLP: u1est = MLP(e_odd || e_even) ----
    {
        int total = M * 128;
        concat_chk<<<(total + 255) / 256, 256, 0, c.s>>>(c.e[lvl], c.X, half, total);
        gemm_wmma<4, true ><<<M / 16, 32, 0, c.s>>>(c.X, c.Pchk1, c.b_chk1, c.H, 128, HDIM);
        gemm_wmma<8, false><<<M / 16, 32, 0, c.s>>>(c.H, c.Pchk2, c.b_chk2, c.e[lvl + 1], HDIM, DEMB);
    }
    decode_rec(c, lvl + 1, bit, half);

    // save left-child results before the right child overwrites level lvl+1
    size_t nb = (size_t)M * sizeof(float);
    hipMemcpyAsync(c.uhat_sv[lvl], c.uhat[lvl + 1], nb, hipMemcpyDeviceToDevice, c.s);
    hipMemcpyAsync(c.llr_sv[lvl],  c.llr[lvl + 1],  nb, hipMemcpyDeviceToDevice, c.s);
    hipMemcpyAsync(c.x_sv[lvl],    c.xb[lvl + 1],   nb, hipMemcpyDeviceToDevice, c.s);
    hipMemcpyAsync(c.ordL_sv[lvl], c.ord_ret[lvl + 1], (size_t)BL * sizeof(int),
                   hipMemcpyDeviceToDevice, c.s);

    // ---- bit-node MLP: u2est = MLP(gather(e_odd,e_even) || emb[u1hp]) ----
    {
        int total = M * 160;
        concat_bit<<<(total + 255) / 256, 256, 0, c.s>>>(c.e[lvl], c.ordL_sv[lvl],
                                                         c.x_sv[lvl], c.emb, c.X, half, total);
        gemm_wmma<5, true ><<<M / 16, 32, 0, c.s>>>(c.X, c.Pbit1, c.b_bit1, c.H, 160, HDIM);
        gemm_wmma<8, false><<<M / 16, 32, 0, c.s>>>(c.H, c.Pbit2, c.b_bit2, c.e[lvl + 1], HDIM, DEMB);
    }
    decode_rec(c, lvl + 1, bit + half, half);

    combine_kernel<<<(M + 255) / 256, 256, 0, c.s>>>(
        half, c.ord_ret[lvl + 1],
        c.uhat_sv[lvl], c.llr_sv[lvl], c.x_sv[lvl], c.ordL_sv[lvl],
        c.uhat[lvl + 1], c.llr[lvl + 1], c.xb[lvl + 1],
        c.uhat[lvl], c.llr[lvl], c.xb[lvl], c.ord_ret[lvl], M);
}

} // namespace

extern "C" void kernel_launch(void* const* d_in, const int* in_sizes, int n_in,
                              void* d_out, int out_size, void* d_ws, size_t ws_size,
                              hipStream_t stream)
{
    (void)in_sizes; (void)n_in; (void)ws_size;
    const float* y      = (const float*)d_in[0];
    const int*   f      = (const int*)  d_in[1];
    const float* W_obs1 = (const float*)d_in[2];
    const float* b_obs1 = (const float*)d_in[3];
    const float* W_obs2 = (const float*)d_in[4];
    const float* b_obs2 = (const float*)d_in[5];
    const float* W_chk1 = (const float*)d_in[6];
    const float* b_chk1 = (const float*)d_in[7];
    const float* W_chk2 = (const float*)d_in[8];
    const float* b_chk2 = (const float*)d_in[9];
    const float* W_bit1 = (const float*)d_in[10];
    const float* b_bit1 = (const float*)d_in[11];
    const float* W_bit2 = (const float*)d_in[12];
    const float* b_bit2 = (const float*)d_in[13];
    const float* W_llr  = (const float*)d_in[14];
    const float* b_llr  = (const float*)d_in[15];
    const float* emb    = (const float*)d_in[16];

    // -------- workspace partition --------
    size_t off = 0;
    auto A = [&](size_t bytes) -> void* {
        void* p = (char*)d_ws + off;
        off = (off + bytes + 255) & ~(size_t)255;
        return p;
    };

    Ctx c{};
    c.s = stream;
    for (int l = 0; l < 7; ++l) {
        int Nl = NBITS >> l;
        c.e[l]    = (float*)A((size_t)BL * Nl * DEMB * sizeof(float));
        c.uhat[l] = (float*)A((size_t)BL * Nl * sizeof(float));
        c.xb[l]   = (float*)A((size_t)BL * Nl * sizeof(float));
        c.llr[l]  = (float*)A((size_t)BL * Nl * sizeof(float));
        c.ord_ret[l] = (int*)A((size_t)BL * sizeof(int));
    }
    for (int l = 0; l < 6; ++l) {
        int half = (NBITS >> l) / 2;
        c.uhat_sv[l] = (float*)A((size_t)BL * half * sizeof(float));
        c.llr_sv[l]  = (float*)A((size_t)BL * half * sizeof(float));
        c.x_sv[l]    = (float*)A((size_t)BL * half * sizeof(float));
        c.ordL_sv[l] = (int*)  A((size_t)BL * sizeof(int));
    }
    c.pm = (float*)A((size_t)BL * sizeof(float));
    const int MMAX = BL * 32;                                     // 32768 rows
    c.X = (_Float16*)A((size_t)MMAX * 160 * sizeof(_Float16));
    c.H = (_Float16*)A((size_t)MMAX * HDIM * sizeof(_Float16));
    float* e0tmp = (float*)A((size_t)BATCH * NBITS * DEMB * sizeof(float));
    c.Pchk1 = (_Float16*)A((size_t)4 * 16 * 512 * sizeof(_Float16));
    c.Pbit1 = (_Float16*)A((size_t)5 * 16 * 512 * sizeof(_Float16));
    c.Pchk2 = (_Float16*)A((size_t)8 * 4 * 512 * sizeof(_Float16));
    c.Pbit2 = (_Float16*)A((size_t)8 * 4 * 512 * sizeof(_Float16));
    c.Pobs2 = (_Float16*)A((size_t)8 * 4 * 512 * sizeof(_Float16));

    c.b_chk1 = b_chk1; c.b_chk2 = b_chk2;
    c.b_bit1 = b_bit1; c.b_bit2 = b_bit2;
    c.emb = emb; c.Wllr = W_llr; c.bllr = b_llr; c.f = f;

    // -------- weight packing (once per call; deterministic) --------
    pack_b<<<(4 * 16 * 512 + 255) / 256, 256, 0, stream>>>(W_chk1, c.Pchk1, 128, 4, 16, HDIM);
    pack_b<<<(5 * 16 * 512 + 255) / 256, 256, 0, stream>>>(W_bit1, c.Pbit1, 144, 5, 16, HDIM);
    pack_b<<<(8 * 4 * 512 + 255) / 256, 256, 0, stream>>>(W_chk2, c.Pchk2, 256, 8, 4, DEMB);
    pack_b<<<(8 * 4 * 512 + 255) / 256, 256, 0, stream>>>(W_bit2, c.Pbit2, 256, 8, 4, DEMB);
    pack_b<<<(8 * 4 * 512 + 255) / 256, 256, 0, stream>>>(W_obs2, c.Pobs2, 256, 8, 4, DEMB);

    init_pm<<<(BL + 255) / 256, 256, 0, stream>>>(c.pm);

    // -------- observation MLP: e0 = MLP_obs(y), broadcast over list dim --------
    const int Mobs = BATCH * NBITS; // 8192 rows
    obs_mlp1<<<(Mobs * HDIM + 255) / 256, 256, 0, stream>>>(y, W_obs1, b_obs1, c.H, Mobs);
    gemm_wmma<8, false><<<Mobs / 16, 32, 0, stream>>>(c.H, c.Pobs2, b_obs2, e0tmp, HDIM, DEMB);
    {
        int total = BL * NBITS * DEMB;
        bcast_e0<<<(total + 255) / 256, 256, 0, stream>>>(e0tmp, c.e[0], total);
    }

    // -------- sequential SCL tree (fixed DFS enqueue) --------
    decode_rec(c, 0, 0, NBITS);

    // -------- outputs: best-list uhat then full llr --------
    {
        int total = BATCH * NBITS + BL * NBITS;
        final_kernel<<<(total + 255) / 256, 256, 0, stream>>>(
            c.pm, c.uhat[0], c.llr[0], (float*)d_out, out_size);
    }
}
